// AttentionBlock_50483045598056
// MI455X (gfx1250) — matrix operand order repaired and checked by reading, not compiled, not run on hardware
//
#include <hip/hip_runtime.h>
#include <stdint.h>

#define DEVFN static __device__ __forceinline__

typedef __attribute__((ext_vector_type(16))) __bf16 v16bf;
typedef __attribute__((ext_vector_type(8)))  __bf16 v8bf;
typedef __attribute__((ext_vector_type(8)))  float  v8f;

union V16U { v16bf v; v8bf h[2]; };

// Problem constants
constexpr int Bn  = 16;
constexpr int Cc  = 512;
constexpr int HWn = 1024;   // 32*32
constexpr int Gg  = 32;
constexpr int CPG = 16;     // channels per group
constexpr int NH  = 8;
constexpr int HD  = 64;
constexpr int MT  = Bn * HWn;  // 16384 rows for the big GEMMs
constexpr float EPSv = 1e-5f;

DEVFN uint16_t f2bf(float f) {
  union { float f; uint32_t u; } x{f};
  uint32_t r = x.u + 0x7FFFu + ((x.u >> 16) & 1u);
  return (uint16_t)(r >> 16);
}

DEVFN v8f wmma_bf16(v16bf a, v16bf b, v8f c) {
  return __builtin_amdgcn_wmma_f32_16x16x32_bf16(false, a, false, b, (short)0, c, false, false);
}

// CDNA5 async global->LDS copy, 16 bytes per lane
DEVFN void cp16_async(uint16_t* lds_dst, const uint16_t* gsrc) {
  unsigned loff = (unsigned)(uintptr_t)lds_dst;       // low 32 bits = LDS offset
  unsigned long long ga = (unsigned long long)(uintptr_t)gsrc;
  asm volatile("global_load_async_to_lds_b128 %0, %1, off"
               :: "v"(loff), "v"(ga) : "memory");
}
DEVFN void async_wait0() {
  asm volatile("s_wait_asynccnt 0" ::: "memory");
}

// ---------------------------------------------------------------------------
// Kernel 0: convert fp32 weights -> bf16
// ---------------------------------------------------------------------------
__global__ void k_convert(const float* __restrict__ wq, const float* __restrict__ wp,
                          uint16_t* __restrict__ wqb, uint16_t* __restrict__ wpb) {
  int i = blockIdx.x * 256 + threadIdx.x;
  int stride = gridDim.x * 256;
  const int nq = 3 * Cc * Cc;
  const int np = Cc * Cc;
  for (int j = i; j < nq; j += stride) wqb[j] = f2bf(wq[j]);
  for (int j = i; j < np; j += stride) wpb[j] = f2bf(wp[j]);
}

// ---------------------------------------------------------------------------
// Kernel 1: GroupNorm, write hn[b*HW + hw][c] in bf16 (transposed for GEMM A).
// Pass 2 packs the group's 16 channels per hw into one 32B store (coalesced).
// ---------------------------------------------------------------------------
__global__ __launch_bounds__(256) void k_gn(const float* __restrict__ x,
                                            const float* __restrict__ gamma,
                                            const float* __restrict__ beta,
                                            uint16_t* __restrict__ hn) {
  __shared__ float red[16];
  __shared__ float mb[2];
  int bg = blockIdx.x;
  int b = bg / Gg, grp = bg % Gg;
  const float* xp = x + ((size_t)b * Cc + (size_t)grp * CPG) * HWn;
  int tid = threadIdx.x;

  float s = 0.f, ss = 0.f;
  for (int e = tid; e < CPG * HWn; e += 256) {
    float v = xp[e];
    s += v; ss += v * v;
  }
  #pragma unroll
  for (int o = 16; o > 0; o >>= 1) {
    s  += __shfl_down(s,  o, 32);
    ss += __shfl_down(ss, o, 32);
  }
  int wv = tid >> 5, ln = tid & 31;
  if (ln == 0) { red[wv] = s; red[8 + wv] = ss; }
  __syncthreads();
  if (tid == 0) {
    float S = 0.f, SS = 0.f;
    #pragma unroll
    for (int i = 0; i < 8; i++) { S += red[i]; SS += red[8 + i]; }
    float inv_n = 1.0f / (float)(CPG * HWn);
    float mean = S * inv_n;
    float var  = SS * inv_n - mean * mean;
    mb[0] = mean;
    mb[1] = rsqrtf(var + EPSv);
  }
  __syncthreads();
  float mean = mb[0], rstd = mb[1];

  float gm[CPG], bt[CPG];
  #pragma unroll
  for (int cc = 0; cc < CPG; cc++) {
    gm[cc] = gamma[grp * CPG + cc] * rstd;
    bt[cc] = beta[grp * CPG + cc];
  }
  for (int hw = tid; hw < HWn; hw += 256) {
    union { uint16_t u[16]; uint4 q4[2]; } pkt;
    #pragma unroll
    for (int cc = 0; cc < CPG; cc++) {
      float v = (xp[cc * HWn + hw] - mean) * gm[cc] + bt[cc];
      pkt.u[cc] = f2bf(v);
    }
    uint4* dst = (uint4*)&hn[((size_t)b * HWn + hw) * Cc + grp * CPG];
    dst[0] = pkt.q4[0];
    dst[1] = pkt.q4[1];
  }
}

// ---------------------------------------------------------------------------
// GEMM skeleton: block tile 128(M)x128(N), 8 waves, each wave 32x64.
// A tile double-buffered in LDS via async copies (issue k+1 while computing k,
// one barrier per K-step). B frags read directly from bf16 [N,K] weights:
// lane n reads 16 contiguous K values (k = 16h + s, same order as the A frags).
// ---------------------------------------------------------------------------
#define BMr 128
#define LDA 40    // padded LDS stride (elements); 80B keeps 16B alignment
#define TP  35    // epilogue transpose pad (floats): conflict-free on 64 banks

// ---------------------------------------------------------------------------
// Kernel 2: QKV GEMM, epilogue scatters q[b,h,i,d], k[b,h,j,d] directly
// (32B-contiguous half-wave chunks) and v[b,h,d,j] via LDS transpose so the
// stores are coalesced along hw.
// ---------------------------------------------------------------------------
__global__ __launch_bounds__(256) void k_qkv(const uint16_t* __restrict__ hn,
                                             const uint16_t* __restrict__ wb,
                                             const float* __restrict__ bias,
                                             uint16_t* __restrict__ qo,
                                             uint16_t* __restrict__ ko,
                                             uint16_t* __restrict__ vo) {
  __shared__ __align__(16) uint16_t At[2][BMr * LDA];
  __shared__ float Tt[8][64 * TP];
  const int m0 = blockIdx.x * BMr;
  const int n0 = blockIdx.y * 128;
  const int tid = threadIdx.x, ln = tid & 31, wv = tid >> 5;
  const int msub = (wv & 3) * 32, nsub = (wv >> 2) * 64;

  auto stage = [&](int buf, int k0) {
    #pragma unroll
    for (int t2 = 0; t2 < 2; t2++) {
      int c = tid * 2 + t2;
      int row = c >> 2, ch = c & 3;
      cp16_async(&At[buf][row * LDA + ch * 8],
                 hn + (size_t)(m0 + row) * Cc + k0 + ch * 8);
    }
  };

  v8f acc[2][4];
  #pragma unroll
  for (int mt = 0; mt < 2; mt++)
    #pragma unroll
    for (int nt = 0; nt < 4; nt++)
      acc[mt][nt] = v8f{0.f, 0.f, 0.f, 0.f, 0.f, 0.f, 0.f, 0.f};

  stage(0, 0);
  constexpr int KS = Cc / 32;
  for (int ks = 0; ks < KS; ks++) {
    const int cur = ks & 1;
    const int k0 = ks * 32;
    async_wait0();
    __syncthreads();
    if (ks + 1 < KS) stage(cur ^ 1, k0 + 32);

    V16U a[2];
    #pragma unroll
    for (int mt = 0; mt < 2; mt++) {
      const uint16_t* p = &At[cur][(msub + mt * 16 + (ln & 15)) * LDA + ((ln >> 4) << 4)];
      a[mt].h[0] = *(const v8bf*)p;
      a[mt].h[1] = *(const v8bf*)(p + 8);
    }
    #pragma unroll
    for (int nt = 0; nt < 4; nt++) {
      int n = n0 + nsub + nt * 16 + (ln & 15);
      v16bf bf = *(const v16bf*)(wb + (size_t)n * Cc + k0 + ((ln >> 4) << 4));
      #pragma unroll
      for (int mt = 0; mt < 2; mt++)
        acc[mt][nt] = wmma_bf16(a[mt].v, bf, acc[mt][nt]);
    }
  }

  // epilogue
  const int gn0 = n0 + nsub;           // 64-aligned -> single (t, head) per wave
  const int t = gn0 >> 9;
  const int head = (gn0 & 511) >> 6;
  const int mbase = m0 + msub;         // 32 rows, all in same batch
  const int b = mbase >> 10, hw0 = mbase & 1023;

  if (t < 2) {
    // q/k: d is the lane axis -> contiguous 32B half-wave stores
    uint16_t* dst = (t == 0) ? qo : ko;
    #pragma unroll
    for (int mt = 0; mt < 2; mt++) {
      #pragma unroll
      for (int nt = 0; nt < 4; nt++) {
        int n = gn0 + nt * 16 + (ln & 15);
        float bv = bias[n];
        int d = n & 63;
        #pragma unroll
        for (int r = 0; r < 8; r++) {
          int m = mbase + mt * 16 + r + ((ln >> 4) << 3);
          int hw = m & 1023;
          dst[(((size_t)b * NH + head) * HWn + hw) * HD + d] =
              f2bf(acc[mt][nt][r] + bv);
        }
      }
    }
  } else {
    // v[b,h,d,hw]: transpose wave tile through LDS, store with lanes along hw
    float* tw = &Tt[wv][0];
    #pragma unroll
    for (int mt = 0; mt < 2; mt++) {
      #pragma unroll
      for (int nt = 0; nt < 4; nt++) {
        int nl = nt * 16 + (ln & 15);
        #pragma unroll
        for (int r = 0; r < 8; r++) {
          int ml = mt * 16 + r + ((ln >> 4) << 3);
          tw[nl * TP + ml] = acc[mt][nt][r];
        }
      }
    }
    size_t basev = ((size_t)b * NH + head) * HD;
    for (int n = 0; n < 64; n++) {
      float val = tw[n * TP + ln] + bias[gn0 + n];
      vo[(basev + n) * HWn + hw0 + ln] = f2bf(val);
    }
  }
}

// ---------------------------------------------------------------------------
// Kernel 3: flash attention. One block = 128 queries of one (b,head);
// each wave owns 16 queries, streams j in tiles of 32 with online softmax.
// ---------------------------------------------------------------------------
__global__ __launch_bounds__(256) void k_attn(const uint16_t* __restrict__ qi,
                                              const uint16_t* __restrict__ ki,
                                              const uint16_t* __restrict__ vi,
                                              uint16_t* __restrict__ ho) {
  __shared__ __align__(16) uint16_t Pt[8][16 * 32];  // per-wave P staging
  int ib = blockIdx.x & 7;
  int h  = (blockIdx.x >> 3) & (NH - 1);
  int b  = blockIdx.x >> 6;
  int tid = threadIdx.x, ln = tid & 31, wv = tid >> 5;
  int i0 = ib * 128 + wv * 16;

  const uint16_t* qb = qi + ((size_t)b * NH + h) * HWn * HD;
  const uint16_t* kb = ki + ((size_t)b * NH + h) * HWn * HD;
  const uint16_t* vb = vi + ((size_t)b * NH + h) * HD * HWn;
  uint16_t* pw = &Pt[wv][0];

  // Q fragments for K(=d) chunks 0..31, 32..63 (resident for whole kernel)
  V16U qf[2];
  {
    int row = i0 + (ln & 15);
    #pragma unroll
    for (int kc = 0; kc < 2; kc++) {
      const uint16_t* p = qb + (size_t)row * HD + kc * 32 + ((ln >> 4) << 4);
      qf[kc].h[0] = *(const v8bf*)p;
      qf[kc].h[1] = *(const v8bf*)(p + 8);
    }
  }

  float mstat[8], lstat[8];
  v8f o[4];
  #pragma unroll
  for (int r = 0; r < 8; r++) { mstat[r] = -3.0e38f; lstat[r] = 0.f; }
  #pragma unroll
  for (int nt = 0; nt < 4; nt++)
    o[nt] = v8f{0.f, 0.f, 0.f, 0.f, 0.f, 0.f, 0.f, 0.f};

  const float scale = 0.125f;  // 1/sqrt(64)

  for (int j0 = 0; j0 < HWn; j0 += 32) {
    // S tiles: S[i, j0+jt*16 .. +15] = q . k
    v8f s[2];
    s[0] = v8f{0.f, 0.f, 0.f, 0.f, 0.f, 0.f, 0.f, 0.f};
    s[1] = s[0];
    #pragma unroll
    for (int jt = 0; jt < 2; jt++) {
      #pragma unroll
      for (int kc = 0; kc < 2; kc++) {
        v16bf bf = *(const v16bf*)(kb + (size_t)(j0 + jt * 16 + (ln & 15)) * HD
                                      + kc * 32 + ((ln >> 4) << 4));
        s[jt] = wmma_bf16(qf[kc].v, bf, s[jt]);
      }
    }

    // online softmax (row = r + 8*(ln>>4); row stats replicated per 16-lane
    // half, reductions are xor-shuffles within the half)
    float p0s[8], p1s[8];
    #pragma unroll
    for (int r = 0; r < 8; r++) {
      float a = s[0][r] * scale;
      float c = s[1][r] * scale;
      float mx = fmaxf(a, c);
      #pragma unroll
      for (int off = 1; off < 16; off <<= 1) mx = fmaxf(mx, __shfl_xor(mx, off, 32));
      float mnew = fmaxf(mstat[r], mx);
      float f = __expf(mstat[r] - mnew);
      float p0 = __expf(a - mnew);
      float p1 = __expf(c - mnew);
      p0s[r] = p0; p1s[r] = p1;
      float rs = p0 + p1;
      #pragma unroll
      for (int off = 1; off < 16; off <<= 1) rs += __shfl_xor(rs, off, 32);
      lstat[r] = lstat[r] * f + rs;
      mstat[r] = mnew;
      #pragma unroll
      for (int nt = 0; nt < 4; nt++) o[nt][r] *= f;
    }

    // stage P (C-layout -> A-layout) through per-wave LDS
    #pragma unroll
    for (int r = 0; r < 8; r++) {
      int rr = r + ((ln >> 4) << 3);
      pw[rr * 32 + (ln & 15)]      = f2bf(p0s[r]);
      pw[rr * 32 + 16 + (ln & 15)] = f2bf(p1s[r]);
    }
    V16U pa;
    {
      const uint16_t* pp = pw + (ln & 15) * 32 + ((ln >> 4) << 4);
      pa.h[0] = *(const v8bf*)pp;
      pa.h[1] = *(const v8bf*)(pp + 8);
    }

    // O += P x V  (B lane n=d reads 16 contiguous j from v[d,:])
    #pragma unroll
    for (int nt = 0; nt < 4; nt++) {
      v16bf bf = *(const v16bf*)(vb + (size_t)(nt * 16 + (ln & 15)) * HWn
                                    + j0 + ((ln >> 4) << 4));
      o[nt] = wmma_bf16(pa.v, bf, o[nt]);
    }
  }

  // finalize: O /= l, write ho[b*HW+i][h*64+d] (32B contiguous per half-wave)
  #pragma unroll
  for (int r = 0; r < 8; r++) {
    float inv = 1.0f / lstat[r];
    int i = i0 + r + ((ln >> 4) << 3);
    #pragma unroll
    for (int nt = 0; nt < 4; nt++) {
      ho[((size_t)b * HWn + i) * Cc + h * HD + nt * 16 + (ln & 15)] =
          f2bf(o[nt][r] * inv);
    }
  }
}

// ---------------------------------------------------------------------------
// Kernel 4: proj GEMM + bias + residual, fp32 output in [B,C,H,W].
// Epilogue transposes each wave tile through LDS so stores (and residual
// reads) are 128B-coalesced along hw.
// ---------------------------------------------------------------------------
__global__ __launch_bounds__(256) void k_proj(const uint16_t* __restrict__ ho,
                                              const uint16_t* __restrict__ wb,
                                              const float* __restrict__ bias,
                                              const float* __restrict__ xin,
                                              float* __restrict__ out) {
  __shared__ __align__(16) uint16_t At[2][BMr * LDA];
  __shared__ float Tt[8][64 * TP];
  const int m0 = blockIdx.x * BMr;
  const int n0 = blockIdx.y * 128;
  const int tid = threadIdx.x, ln = tid & 31, wv = tid >> 5;
  const int msub = (wv & 3) * 32, nsub = (wv >> 2) * 64;

  auto stage = [&](int buf, int k0) {
    #pragma unroll
    for (int t2 = 0; t2 < 2; t2++) {
      int c = tid * 2 + t2;
      int row = c >> 2, ch = c & 3;
      cp16_async(&At[buf][row * LDA + ch * 8],
                 ho + (size_t)(m0 + row) * Cc + k0 + ch * 8);
    }
  };

  v8f acc[2][4];
  #pragma unroll
  for (int mt = 0; mt < 2; mt++)
    #pragma unroll
    for (int nt = 0; nt < 4; nt++)
      acc[mt][nt] = v8f{0.f, 0.f, 0.f, 0.f, 0.f, 0.f, 0.f, 0.f};

  stage(0, 0);
  constexpr int KS = Cc / 32;
  for (int ks = 0; ks < KS; ks++) {
    const int cur = ks & 1;
    const int k0 = ks * 32;
    async_wait0();
    __syncthreads();
    if (ks + 1 < KS) stage(cur ^ 1, k0 + 32);

    V16U a[2];
    #pragma unroll
    for (int mt = 0; mt < 2; mt++) {
      const uint16_t* p = &At[cur][(msub + mt * 16 + (ln & 15)) * LDA + ((ln >> 4) << 4)];
      a[mt].h[0] = *(const v8bf*)p;
      a[mt].h[1] = *(const v8bf*)(p + 8);
    }
    #pragma unroll
    for (int nt = 0; nt < 4; nt++) {
      int n = n0 + nsub + nt * 16 + (ln & 15);
      v16bf bf = *(const v16bf*)(wb + (size_t)n * Cc + k0 + ((ln >> 4) << 4));
      #pragma unroll
      for (int mt = 0; mt < 2; mt++)
        acc[mt][nt] = wmma_bf16(a[mt].v, bf, acc[mt][nt]);
    }
  }

  // epilogue: LDS transpose -> coalesced residual add + store along hw
  float* tw = &Tt[wv][0];
  #pragma unroll
  for (int mt = 0; mt < 2; mt++) {
    #pragma unroll
    for (int nt = 0; nt < 4; nt++) {
      int nl = nt * 16 + (ln & 15);
      #pragma unroll
      for (int r = 0; r < 8; r++) {
        int ml = mt * 16 + r + ((ln >> 4) << 3);
        tw[nl * TP + ml] = acc[mt][nt][r];
      }
    }
  }
  const int mbase = m0 + msub;
  const int b = mbase >> 10, hw0 = mbase & 1023;
  for (int n = 0; n < 64; n++) {
    int gn = n0 + nsub + n;
    float val = tw[n * TP + ln] + bias[gn];
    size_t idx = ((size_t)b * Cc + gn) * HWn + hw0 + ln;
    out[idx] = xin[idx] + val;
  }
}

// ---------------------------------------------------------------------------
extern "C" void kernel_launch(void* const* d_in, const int* in_sizes, int n_in,
                              void* d_out, int out_size, void* d_ws, size_t ws_size,
                              hipStream_t stream) {
  const float* x     = (const float*)d_in[0];
  const float* gamma = (const float*)d_in[1];
  const float* beta  = (const float*)d_in[2];
  const float* wqkv  = (const float*)d_in[3];
  const float* bqkv  = (const float*)d_in[4];
  const float* wproj = (const float*)d_in[5];
  const float* bproj = (const float*)d_in[6];
  float* out = (float*)d_out;

  char* ws = (char*)d_ws;
  size_t off = 0;
  auto alloc = [&](size_t bytes) -> void* {
    void* p = ws + off;
    off += (bytes + 255) & ~(size_t)255;
    return p;
  };
  uint16_t* hn  = (uint16_t*)alloc((size_t)MT * Cc * 2);           // normed x, [M,K]
  uint16_t* q   = (uint16_t*)alloc((size_t)Bn * NH * HWn * HD * 2);
  uint16_t* k   = (uint16_t*)alloc((size_t)Bn * NH * HWn * HD * 2);
  uint16_t* v   = (uint16_t*)alloc((size_t)Bn * NH * HD * HWn * 2);
  uint16_t* ho  = (uint16_t*)alloc((size_t)MT * Cc * 2);           // attn out, [M,K]
  uint16_t* wqb = (uint16_t*)alloc((size_t)3 * Cc * Cc * 2);
  uint16_t* wpb = (uint16_t*)alloc((size_t)Cc * Cc * 2);

  k_convert<<<512, 256, 0, stream>>>(wqkv, wproj, wqb, wpb);
  k_gn<<<Bn * Gg, 256, 0, stream>>>(x, gamma, beta, hn);
  k_qkv<<<dim3(MT / BMr, (3 * Cc) / 128), 256, 0, stream>>>(hn, wqb, bqkv, q, k, v);
  k_attn<<<Bn * NH * (HWn / 128), 256, 0, stream>>>(q, k, v, ho);
  k_proj<<<dim3(MT / BMr, Cc / 128), 256, 0, stream>>>(ho, wpb, bproj, x, out);
}
